// SpatioTemporalGNN_29420525978025
// MI455X (gfx1250) — compile-verified, matrix operand order checked by reading, not run on hardware
//
#include <hip/hip_runtime.h>

#define BB   4
#define NN   2000
#define TT   24
#define FIN  16
#define HH   128
#define EE   64000
#define CC   6
#define BN   (BB*NN)     // 8000
#define G3H  (3*HH)      // 384

typedef __attribute__((ext_vector_type(16))) _Float16 v16h;
typedef __attribute__((ext_vector_type(8)))  _Float16 v8h;
typedef __attribute__((ext_vector_type(8)))  float    v8f;

// ---------------------------------------------------------------------------
// WMMA helpers (CDNA5 v_wmma_f32_16x16x32_f16, wave32)
// A 16x32 f16 operand: lane L holds row m=L%16, half=L/16; its 16 halves are
// two contiguous 8-half chunks at k0+half*8 and k0+16+half*8 (ISA 7.12.2).
// B 32x16 f16 operand is the mirror: lane holds column n=L%16 -> a row of W
// when computing X @ W^T. Same chunk pattern, so one loader serves both.
// ---------------------------------------------------------------------------
__device__ __forceinline__ v16h load_frag(const _Float16* __restrict__ row, int k0, int half) {
    v8h lo = *(const v8h*)(row + k0 + half * 8);
    v8h hi = *(const v8h*)(row + k0 + 16 + half * 8);
    v16h r;
#pragma unroll
    for (int e = 0; e < 8; ++e) { r[e] = lo[e]; r[8 + e] = hi[e]; }
    return r;
}

__device__ __forceinline__ v8f wmma16(v16h a, v16h b, v8f c) {
    return __builtin_amdgcn_wmma_f32_16x16x32_f16(false, a, false, b, (short)0, c, false, false);
}

__device__ __forceinline__ v8f zero8() {
    v8f z;
#pragma unroll
    for (int v = 0; v < 8; ++v) z[v] = 0.f;
    return z;
}

__device__ __forceinline__ float sigmoid_f(float x) { return 1.0f / (1.0f + __expf(-x)); }
__device__ __forceinline__ float tanh_f(float x) {
    x = fminf(15.0f, fmaxf(-15.0f, x));
    float e = __expf(2.0f * x);
    return (e - 1.0f) / (e + 1.0f);
}

// ---------------------------------------------------------------------------
// Weight conversion fp32 -> f16 (W_ih padded K 16->32, cls_W2 padded N 6->16)
// ---------------------------------------------------------------------------
#define WHH_N (G3H*HH)        // 49152
#define WIH_N (G3H*32)        // 12288
#define GW_N  (HH*HH)         // 16384
#define CW1_N (HH*2*HH)       // 32768
#define CW2_N (16*HH)         // 2048
#define CONV_TOTAL (WHH_N + WIH_N + 2*GW_N + CW1_N + CW2_N)  // 129024

__global__ __launch_bounds__(256) void convert_weights_kernel(
    const float* __restrict__ Wih, const float* __restrict__ Whh,
    const float* __restrict__ gW1, const float* __restrict__ gW2,
    const float* __restrict__ cW1, const float* __restrict__ cW2,
    _Float16* __restrict__ whh16, _Float16* __restrict__ wih16,
    _Float16* __restrict__ gw116, _Float16* __restrict__ gw216,
    _Float16* __restrict__ cw116, _Float16* __restrict__ cw216) {
    int i = blockIdx.x * 256 + threadIdx.x;
    if (i >= CONV_TOTAL) return;
    if (i < WHH_N) {
        whh16[i] = (_Float16)Whh[i];
    } else if (i < WHH_N + WIH_N) {
        int j = i - WHH_N, n = j >> 5, k = j & 31;
        wih16[j] = (k < FIN) ? (_Float16)Wih[n * FIN + k] : (_Float16)0.f;
    } else if (i < WHH_N + WIH_N + GW_N) {
        int j = i - (WHH_N + WIH_N);
        gw116[j] = (_Float16)gW1[j];
    } else if (i < WHH_N + WIH_N + 2 * GW_N) {
        int j = i - (WHH_N + WIH_N + GW_N);
        gw216[j] = (_Float16)gW2[j];
    } else if (i < WHH_N + WIH_N + 2 * GW_N + CW1_N) {
        int j = i - (WHH_N + WIH_N + 2 * GW_N);
        cw116[j] = (_Float16)cW1[j];
    } else {
        int j = i - (WHH_N + WIH_N + 2 * GW_N + CW1_N);
        int n = j >> 7, k = j & 127;
        cw216[j] = (n < CC) ? (_Float16)cW2[n * HH + k] : (_Float16)0.f;
    }
}

__global__ __launch_bounds__(256) void zero_f32_kernel(float* __restrict__ p, int n) {
    int i = blockIdx.x * 256 + threadIdx.x;
    if (i < n) p[i] = 0.f;
}

__global__ __launch_bounds__(256) void deg_count_kernel(const int* __restrict__ ei,
                                                        float* __restrict__ deg) {
    int e = blockIdx.x * 256 + threadIdx.x;
    if (e >= EE) return;
    int d = ei[EE + e];
    d = d < 0 ? 0 : (d >= NN ? NN - 1 : d);
    atomicAdd(&deg[d], 1.0f);
}

__global__ __launch_bounds__(256) void deg_inv_kernel(const float* __restrict__ deg,
                                                      float* __restrict__ dinv) {
    int n = blockIdx.x * 256 + threadIdx.x;
    if (n < NN) dinv[n] = rsqrtf(fmaxf(deg[n], 1.0f));
}

// ---------------------------------------------------------------------------
// GRU: one block = 16 sequences, 8 waves. Each wave owns gate columns
// [wave*16, wave*16+16) of r, z and n -> straight-line code, no divergence
// around WMMA. All 24 x-timesteps staged to LDS once (24 KB as padded f16);
// weight B-fragments are loop-invariant and get hoisted to VGPRs.
// Recurrence: 15 WMMAs/wave/step, 2 barriers/step.
// ---------------------------------------------------------------------------
__global__ __launch_bounds__(256) void gru_kernel(
    const float* __restrict__ x,          // [BN, T, FIN]
    const _Float16* __restrict__ wih,     // [384, 32] padded
    const _Float16* __restrict__ whh,     // [384, 128]
    const float* __restrict__ bih, const float* __restrict__ bhh,
    float* __restrict__ temporal) {       // [BN, 128]
    __shared__ _Float16 xall[TT * 16 * 32];   // 24 KB: [t][m][k padded 16->32]
    __shared__ _Float16 h16[16 * 128];        // 4 KB
    __shared__ float    h32[16 * 128];        // 8 KB
    __shared__ float    pre[16 * 512];        // 32 KB: [r|z|nh|ni]

    const int tid  = threadIdx.x;
    const int lane = tid & 31;
    const int l16  = lane & 15;
    const int half = lane >> 4;
    const int wave = tid >> 5;
    const int s0   = blockIdx.x * 16;

    // stage all timesteps: 16 seq x 24 t x 16 f contiguous floats
    {
        const float* xbase = x + (size_t)s0 * TT * FIN;
#pragma unroll
        for (int j = 0; j < 24; ++j) {
            int i = tid + 256 * j;            // 0..6143
            int m = i / (TT * FIN);
            int rem = i - m * (TT * FIN);
            int t = rem >> 4, f = rem & 15;
            xall[(t * 16 + m) * 32 + f]      = (_Float16)xbase[i];
            xall[(t * 16 + m) * 32 + 16 + f] = (_Float16)0.f;   // K-pad
        }
    }
#pragma unroll
    for (int j = 0; j < 8; ++j) {
        int i = tid + 256 * j;
        h16[i] = (_Float16)0.f;
        h32[i] = 0.f;
    }
    __syncthreads();

    const int nr = wave * 16 + l16;           // r column; z = nr+128, n = nr+256
    const _Float16* wr = whh + nr * 128;
    const _Float16* wz = whh + (nr + 128) * 128;
    const _Float16* wn = whh + (nr + 256) * 128;
    const _Float16* xr = wih + nr * 32;
    const _Float16* xz = wih + (nr + 128) * 32;
    const _Float16* xn = wih + (nr + 256) * 32;

    for (int t = 0; t < TT; ++t) {
        // A fragments: h (4 chunks) + x_t (1 chunk), shared by all 3 gates
        v16h a0 = load_frag(h16 + l16 * 128, 0, half);
        v16h a1 = load_frag(h16 + l16 * 128, 32, half);
        v16h a2 = load_frag(h16 + l16 * 128, 64, half);
        v16h a3 = load_frag(h16 + l16 * 128, 96, half);
        v16h ax = load_frag(xall + (t * 16 + l16) * 32, 0, half);

        // r gate (input proj folded in)
        {
            v8f acc = zero8();
            acc = wmma16(a0, load_frag(wr, 0, half), acc);
            acc = wmma16(a1, load_frag(wr, 32, half), acc);
            acc = wmma16(a2, load_frag(wr, 64, half), acc);
            acc = wmma16(a3, load_frag(wr, 96, half), acc);
            acc = wmma16(ax, load_frag(xr, 0, half), acc);
#pragma unroll
            for (int v = 0; v < 8; ++v) pre[(v + 8 * half) * 512 + nr] = acc[v];
        }
        // z gate
        {
            v8f acc = zero8();
            acc = wmma16(a0, load_frag(wz, 0, half), acc);
            acc = wmma16(a1, load_frag(wz, 32, half), acc);
            acc = wmma16(a2, load_frag(wz, 64, half), acc);
            acc = wmma16(a3, load_frag(wz, 96, half), acc);
            acc = wmma16(ax, load_frag(xz, 0, half), acc);
#pragma unroll
            for (int v = 0; v < 8; ++v) pre[(v + 8 * half) * 512 + 128 + nr] = acc[v];
        }
        // n gate: hidden and input parts kept separate (n = tanh(ni + r*nh))
        {
            v8f acc = zero8();
            acc = wmma16(a0, load_frag(wn, 0, half), acc);
            acc = wmma16(a1, load_frag(wn, 32, half), acc);
            acc = wmma16(a2, load_frag(wn, 64, half), acc);
            acc = wmma16(a3, load_frag(wn, 96, half), acc);
#pragma unroll
            for (int v = 0; v < 8; ++v) pre[(v + 8 * half) * 512 + 256 + nr] = acc[v];
            v8f acci = zero8();
            acci = wmma16(ax, load_frag(xn, 0, half), acci);
#pragma unroll
            for (int v = 0; v < 8; ++v) pre[(v + 8 * half) * 512 + 384 + nr] = acci[v];
        }
        __syncthreads();

        // gate math: 2048 elements / 256 threads
#pragma unroll
        for (int j = 0; j < 8; ++j) {
            int i = tid * 8 + j;
            int m = i >> 7, c = i & 127;
            float pr = pre[m * 512 + c]       + bih[c]       + bhh[c];
            float pz = pre[m * 512 + 128 + c] + bih[128 + c] + bhh[128 + c];
            float nh = pre[m * 512 + 256 + c] + bhh[256 + c];
            float ni = pre[m * 512 + 384 + c] + bih[256 + c];
            float r = sigmoid_f(pr);
            float z = sigmoid_f(pz);
            float nn2 = tanh_f(ni + r * nh);
            float hv = (1.f - z) * nn2 + z * h32[m * 128 + c];
            h32[m * 128 + c] = hv;
            h16[m * 128 + c] = (_Float16)hv;
        }
        __syncthreads();
    }

#pragma unroll
    for (int j = 0; j < 8; ++j) {
        int i = tid + 256 * j;
        temporal[(size_t)(s0 + (i >> 7)) * HH + (i & 127)] = h32[i];
    }
}

// ---------------------------------------------------------------------------
// Normalized-adjacency scatter: one wave per (batch, edge); float4 per lane
// covers H=128; accumulate with global fp32 atomics.
// ---------------------------------------------------------------------------
__global__ __launch_bounds__(128) void scatter_kernel(
    const float* __restrict__ x, const int* __restrict__ ei,
    const float* __restrict__ dinv, float* __restrict__ agg) {
    int gw = blockIdx.x * 4 + (threadIdx.x >> 5);   // 0 .. BB*EE-1
    int lane = threadIdx.x & 31;
    int b = gw / EE;
    int e = gw - b * EE;
    int src = ei[e];
    int dst = ei[EE + e];
    src = src < 0 ? 0 : (src >= NN ? NN - 1 : src);
    dst = dst < 0 ? 0 : (dst >= NN ? NN - 1 : dst);
    float norm = dinv[src] * dinv[dst];
    float4 v = ((const float4*)(x + ((size_t)b * NN + src) * HH))[lane];
    float* ad = agg + ((size_t)b * NN + dst) * HH + lane * 4;
    atomicAdd(ad + 0, v.x * norm);
    atomicAdd(ad + 1, v.y * norm);
    atomicAdd(ad + 2, v.z * norm);
    atomicAdd(ad + 3, v.w * norm);
}

// ---------------------------------------------------------------------------
// Fused GCN linear (WMMA) + bias + LayerNorm + ReLU. 16 rows per block,
// 4 waves x 2 tiles cover the 128 output columns.
// ---------------------------------------------------------------------------
__global__ __launch_bounds__(128) void gcn_lin_ln_relu_kernel(
    const float* __restrict__ in, const _Float16* __restrict__ w,
    const float* __restrict__ bias, const float* __restrict__ gam,
    const float* __restrict__ bet, float* __restrict__ out) {
    __shared__ _Float16 inb[16 * 128];
    __shared__ float    outb[16 * 128];
    __shared__ float    red[128 * 2];
    __shared__ float    stats[16 * 2];
    const int tid = threadIdx.x, lane = tid & 31, l16 = lane & 15;
    const int half = lane >> 4, wave = tid >> 5;
    const int r0 = blockIdx.x * 16;

#pragma unroll
    for (int j = 0; j < 16; ++j) {
        int i = tid + 128 * j;
        inb[i] = (_Float16)in[(size_t)(r0 + (i >> 7)) * HH + (i & 127)];
    }
    __syncthreads();

#pragma unroll
    for (int i = 0; i < 2; ++i) {
        int tile = wave * 2 + i;
        int n = tile * 16 + l16;
        v8f acc = zero8();
#pragma unroll
        for (int k0 = 0; k0 < HH; k0 += 32)
            acc = wmma16(load_frag(inb + l16 * 128, k0, half),
                         load_frag(w + n * 128, k0, half), acc);
#pragma unroll
        for (int v = 0; v < 8; ++v) outb[(v + 8 * half) * 128 + n] = acc[v] + bias[n];
    }
    __syncthreads();

    {   // LayerNorm partials: thread (m = tid&15, chunk = tid>>4) sums 16 elems
        int m = tid & 15, ch = tid >> 4;
        float s = 0.f, s2 = 0.f;
#pragma unroll
        for (int j = 0; j < 16; ++j) {
            float v = outb[m * 128 + ch * 16 + j];
            s += v; s2 += v * v;
        }
        red[(m * 8 + ch) * 2 + 0] = s;
        red[(m * 8 + ch) * 2 + 1] = s2;
    }
    __syncthreads();
    if (tid < 16) {
        float s = 0.f, s2 = 0.f;
#pragma unroll
        for (int ch = 0; ch < 8; ++ch) {
            s += red[(tid * 8 + ch) * 2 + 0];
            s2 += red[(tid * 8 + ch) * 2 + 1];
        }
        float mean = s * (1.0f / 128.0f);
        float var = s2 * (1.0f / 128.0f) - mean * mean;
        stats[tid * 2 + 0] = mean;
        stats[tid * 2 + 1] = rsqrtf(var + 1e-5f);
    }
    __syncthreads();
#pragma unroll
    for (int j = 0; j < 16; ++j) {
        int i = tid + 128 * j;
        int m = i >> 7, c = i & 127;
        float v = (outb[m * 128 + c] - stats[m * 2]) * stats[m * 2 + 1] * gam[c] + bet[c];
        out[(size_t)(r0 + m) * HH + c] = fmaxf(v, 0.f);
    }
}

// ---------------------------------------------------------------------------
// Classifier: concat [temporal|s] -> 256, WMMA GEMM -> ReLU -> f16, then one
// padded 128->16 WMMA tile; first 6 columns of each row go to d_out.
// ---------------------------------------------------------------------------
__global__ __launch_bounds__(128) void classifier_kernel(
    const float* __restrict__ temporal, const float* __restrict__ s,
    const _Float16* __restrict__ w1, const float* __restrict__ b1,
    const _Float16* __restrict__ w2, const float* __restrict__ b2,
    float* __restrict__ out) {
    __shared__ _Float16 comb[16 * 256];
    __shared__ _Float16 hb[16 * 128];
    const int tid = threadIdx.x, lane = tid & 31, l16 = lane & 15;
    const int half = lane >> 4, wave = tid >> 5;
    const int r0 = blockIdx.x * 16;

#pragma unroll
    for (int j = 0; j < 32; ++j) {
        int i = tid + 128 * j;
        int m = i >> 8, c = i & 255;
        float v = (c < HH) ? temporal[(size_t)(r0 + m) * HH + c]
                           : s[(size_t)(r0 + m) * HH + (c - HH)];
        comb[i] = (_Float16)v;
    }
    __syncthreads();

#pragma unroll
    for (int i = 0; i < 2; ++i) {
        int tile = wave * 2 + i;
        int n = tile * 16 + l16;
        v8f acc = zero8();
#pragma unroll
        for (int k0 = 0; k0 < 2 * HH; k0 += 32)
            acc = wmma16(load_frag(comb + l16 * 256, k0, half),
                         load_frag(w1 + n * 256, k0, half), acc);
#pragma unroll
        for (int v = 0; v < 8; ++v)
            hb[(v + 8 * half) * 128 + n] = (_Float16)fmaxf(acc[v] + b1[n], 0.f);
    }
    __syncthreads();

    // scalarized so the WMMA tail sits behind an s_cbranch, not EXEC masking
    if (__builtin_amdgcn_readfirstlane(wave) == 0) {
        v8f acc = zero8();
#pragma unroll
        for (int k0 = 0; k0 < HH; k0 += 32)
            acc = wmma16(load_frag(hb + l16 * 128, k0, half),
                         load_frag(w2 + l16 * 128, k0, half), acc);
        if (l16 < CC) {
#pragma unroll
            for (int v = 0; v < 8; ++v)
                out[(size_t)(r0 + v + 8 * half) * CC + l16] = acc[v] + b2[l16];
        }
    }
}

// ---------------------------------------------------------------------------
// Host side
// ---------------------------------------------------------------------------
extern "C" void kernel_launch(void* const* d_in, const int* in_sizes, int n_in,
                              void* d_out, int out_size, void* d_ws, size_t ws_size,
                              hipStream_t stream) {
    (void)in_sizes; (void)n_in; (void)out_size; (void)ws_size;
    const float* node_features = (const float*)d_in[0];
    const int*   edge_index    = (const int*)d_in[1];
    const float* W_ih   = (const float*)d_in[2];
    const float* W_hh   = (const float*)d_in[3];
    const float* b_ih   = (const float*)d_in[4];
    const float* b_hh   = (const float*)d_in[5];
    const float* gcn_W1 = (const float*)d_in[6];
    const float* gcn_b1 = (const float*)d_in[7];
    const float* gcn_W2 = (const float*)d_in[8];
    const float* gcn_b2 = (const float*)d_in[9];
    const float* ln_g1  = (const float*)d_in[10];
    const float* ln_b1  = (const float*)d_in[11];
    const float* ln_g2  = (const float*)d_in[12];
    const float* ln_b2  = (const float*)d_in[13];
    const float* cls_W1 = (const float*)d_in[14];
    const float* cls_b1 = (const float*)d_in[15];
    const float* cls_W2 = (const float*)d_in[16];
    const float* cls_b2 = (const float*)d_in[17];
    float* out = (float*)d_out;

    // workspace layout
    float* temporal = (float*)d_ws;                 // BN*HH f32
    float* agg      = temporal + (size_t)BN * HH;
    float* sbuf     = agg + (size_t)BN * HH;
    float* deg      = sbuf + (size_t)BN * HH;       // 2048 f32
    float* dinv     = deg + 2048;                   // 2048 f32
    _Float16* whh16 = (_Float16*)(dinv + 2048);     // 49152
    _Float16* wih16 = whh16 + WHH_N;                // 12288
    _Float16* gw116 = wih16 + WIH_N;                // 16384
    _Float16* gw216 = gw116 + GW_N;                 // 16384
    _Float16* cw116 = gw216 + GW_N;                 // 32768
    _Float16* cw216 = cw116 + CW1_N;                // 2048

    // 1) weight conversion to f16
    convert_weights_kernel<<<(CONV_TOTAL + 255) / 256, 256, 0, stream>>>(
        W_ih, W_hh, gcn_W1, gcn_W2, cls_W1, cls_W2,
        whh16, wih16, gw116, gw216, cw116, cw216);

    // 2) degree normalization
    zero_f32_kernel<<<8, 256, 0, stream>>>(deg, 2048);
    deg_count_kernel<<<(EE + 255) / 256, 256, 0, stream>>>(edge_index, deg);
    deg_inv_kernel<<<(NN + 255) / 256, 256, 0, stream>>>(deg, dinv);

    // 3) GRU over time -> temporal [BN,128]
    gru_kernel<<<BN / 16, 256, 0, stream>>>(node_features, wih16, whh16, b_ih, b_hh, temporal);

    // 4) GCN layer 1
    zero_f32_kernel<<<(BN * HH + 255) / 256, 256, 0, stream>>>(agg, BN * HH);
    scatter_kernel<<<BB * EE / 4, 128, 0, stream>>>(temporal, edge_index, dinv, agg);
    gcn_lin_ln_relu_kernel<<<BN / 16, 128, 0, stream>>>(agg, gw116, gcn_b1, ln_g1, ln_b1, sbuf);

    // 5) GCN layer 2
    zero_f32_kernel<<<(BN * HH + 255) / 256, 256, 0, stream>>>(agg, BN * HH);
    scatter_kernel<<<BB * EE / 4, 128, 0, stream>>>(sbuf, edge_index, dinv, agg);
    gcn_lin_ln_relu_kernel<<<BN / 16, 128, 0, stream>>>(agg, gw216, gcn_b2, ln_g2, ln_b2, sbuf);

    // 6) classifier -> logits [BN, 6]
    classifier_kernel<<<BN / 16, 128, 0, stream>>>(temporal, sbuf, cw116, cls_b1, cw216, cls_b2, out);
}
